// Attention_17222818857675
// MI455X (gfx1250) — matrix operand order repaired and checked by reading, not compiled, not run on hardware
//
#include <hip/hip_runtime.h>

typedef __attribute__((ext_vector_type(16))) __bf16 v16bf;
typedef __attribute__((ext_vector_type(8)))  __bf16 v8bf;
typedef __attribute__((ext_vector_type(4)))  __bf16 v4bf;
typedef __attribute__((ext_vector_type(8)))  float  v8f;
typedef __attribute__((ext_vector_type(4)))  int    v4i;

constexpr int B_  = 4;
constexpr int N_  = 2048;
constexpr int C_  = 1152;
constexpr int H_  = 16;
constexpr int DH_ = 72;
constexpr int DP_ = 96;            // DH padded to 3*32 for the WMMA K loop
constexpr int CO_ = 3 * C_;        // 3456
constexpr int M_  = B_ * N_;       // 8192

// --------------------------- CDNA5 async copy ------------------------------
#if defined(__AMDGCN__) && __has_builtin(__builtin_amdgcn_global_load_async_to_lds_b128)
#define ASYNC_COPY 1
#else
#define ASYNC_COPY 0
#endif

#if ASYNC_COPY
static __device__ __forceinline__ void async_b128(const __bf16* g, __bf16* l) {
  __builtin_amdgcn_global_load_async_to_lds_b128(
      (__attribute__((address_space(1))) v4i*)g,
      (__attribute__((address_space(3))) v4i*)l, 0, 0);
}
static __device__ __forceinline__ void async_wait0() {
#if __has_builtin(__builtin_amdgcn_s_wait_asynccnt)
  __builtin_amdgcn_s_wait_asynccnt(0);
#else
  asm volatile("s_wait_asynccnt 0" ::: "memory");
#endif
}
#endif

static __device__ __forceinline__ v8f vzero8() {
  v8f z;
#pragma unroll
  for (int i = 0; i < 8; ++i) z[i] = 0.0f;
  return z;
}

// D = A*B + C, bf16 operands, f32 accumulate
static __device__ __forceinline__ v8f wmma_bf16(v16bf a, v16bf b, v8f c) {
  return __builtin_amdgcn_wmma_f32_16x16x32_bf16(false, a, false, b,
                                                 (short)0, c, false, false);
}

// A fragment (16x32 bf16): lane half h reads k = {8h..8h+7} and {16+8h..23+8h}
static __device__ __forceinline__ v16bf frag_a(const __bf16* rowptr, int half) {
  v8bf lo = *(const v8bf*)(rowptr + 8 * half);
  v8bf hi = *(const v8bf*)(rowptr + 16 + 8 * half);
  v16bf r;
#pragma unroll
  for (int i = 0; i < 8; ++i) { r[i] = lo[i]; r[8 + i] = hi[i]; }
  return r;
}

// B fragment (32x16 bf16): lane (col = lane%16, half) reads k = {8h..8h+7} and {16+8h..23+8h}
static __device__ __forceinline__ v16bf frag_b(const __bf16* colptr, int half) {
  v8bf lo = *(const v8bf*)(colptr + 8 * half);
  v8bf hi = *(const v8bf*)(colptr + 16 + 8 * half);
  v16bf r;
#pragma unroll
  for (int i = 0; i < 8; ++i) { r[i] = lo[i]; r[8 + i] = hi[i]; }
  return r;
}

// ---------------------------------------------------------------------------
// Kernel 1: qkv = x @ w_qkv^T  -> scatter bf16 into padded [3,B,H,N,DP]
// ---------------------------------------------------------------------------
__global__ __launch_bounds__(256)
void k_qkv_gemm(const float* __restrict__ x, const float* __restrict__ wqkv,
                __bf16* __restrict__ qkv) {
  __shared__ __bf16 sA[64][40];
  __shared__ __bf16 sB[128][40];
  const int t = threadIdx.x;
  const int m0 = blockIdx.x * 64;
  const int n0 = blockIdx.y * 128;
  const int w = t >> 5, lane = t & 31;
  const int half = lane >> 4, l16 = lane & 15;
  const int mw = (w & 1) * 32, nw = (w >> 1) * 32;

  v8f acc[2][2];
  acc[0][0] = vzero8(); acc[0][1] = vzero8();
  acc[1][0] = vzero8(); acc[1][1] = vzero8();

  for (int kt = 0; kt < C_; kt += 32) {
#pragma unroll
    for (int it = 0; it < 2; ++it) {           // A: 64x32 f32 -> bf16
      int idx = t + it * 256;
      int row = idx >> 3, c4 = (idx & 7) * 4;
      float4 f = *(const float4*)(x + (size_t)(m0 + row) * C_ + kt + c4);
      v4bf v; v[0] = (__bf16)f.x; v[1] = (__bf16)f.y;
      v[2] = (__bf16)f.z; v[3] = (__bf16)f.w;
      *(v4bf*)&sA[row][c4] = v;
    }
#pragma unroll
    for (int it = 0; it < 4; ++it) {           // B: 128x32 f32 -> bf16
      int idx = t + it * 256;
      int row = idx >> 3, c4 = (idx & 7) * 4;
      float4 f = *(const float4*)(wqkv + (size_t)(n0 + row) * C_ + kt + c4);
      v4bf v; v[0] = (__bf16)f.x; v[1] = (__bf16)f.y;
      v[2] = (__bf16)f.z; v[3] = (__bf16)f.w;
      *(v4bf*)&sB[row][c4] = v;
    }
    if (kt + 32 < C_) {                        // prefetch next k-tile
      __builtin_prefetch(x    + (size_t)(m0 + (t >> 3)) * C_ + kt + 32, 0, 0);
      __builtin_prefetch(wqkv + (size_t)(n0 + (t >> 3)) * C_ + kt + 32, 0, 0);
    }
    __syncthreads();
    v16bf a0 = frag_a(&sA[mw + l16][0], half);
    v16bf a1 = frag_a(&sA[mw + 16 + l16][0], half);
    v16bf b0 = frag_b(&sB[nw + l16][0], half);
    v16bf b1 = frag_b(&sB[nw + 16 + l16][0], half);
    acc[0][0] = wmma_bf16(a0, b0, acc[0][0]);
    acc[0][1] = wmma_bf16(a0, b1, acc[0][1]);
    acc[1][0] = wmma_bf16(a1, b0, acc[1][0]);
    acc[1][1] = wmma_bf16(a1, b1, acc[1][1]);
    __syncthreads();
  }

  // scatter: column decomposition hoisted out of the per-element loops
#pragma unroll
  for (int j = 0; j < 2; ++j) {
    int d = n0 + nw + j * 16 + l16;
    int which = d / C_, rem = d - which * C_;
    int hh = rem / DH_, dh = rem - hh * DH_;
    size_t colbase = ((size_t)which * B_ * H_ + hh) * ((size_t)N_ * DP_) + dh;
#pragma unroll
    for (int i = 0; i < 2; ++i)
#pragma unroll
      for (int r = 0; r < 8; ++r) {
        int gm = m0 + mw + i * 16 + r + 8 * half;
        int bb = gm >> 11, nn = gm & (N_ - 1);
        qkv[colbase + ((size_t)bb * H_ * N_ + nn) * DP_] = (__bf16)acc[i][j][r];
      }
  }
}

// ---------------------------------------------------------------------------
// Kernel 2: RoPE + RMS-norm on q and k rows, in place (16 rows per wave)
// ---------------------------------------------------------------------------
__global__ __launch_bounds__(256)
void k_rope_rms(__bf16* __restrict__ qkv, const float* __restrict__ qnw,
                const float* __restrict__ knw) {
  const int w = threadIdx.x >> 5, lane = threadIdx.x & 31;
  const int gw = blockIdx.x * 8 + w;          // 16384 waves
  // per-lane RoPE frequencies (independent of the row)
  float fr[2];
  fr[0] = __powf(10000.0f, -(float)lane * (1.0f / 36.0f));
  fr[1] = __powf(10000.0f, -(float)(lane + 32) * (1.0f / 36.0f));
  for (int rr = 0; rr < 16; ++rr) {
    int ri = gw * 16 + rr;                    // 0 .. 2*B*H*N-1
    int which = ri >> 17;                     // B*H*N = 2^17
    int rem = ri & ((1 << 17) - 1);
    int b = rem >> 15, h = (rem >> 11) & 15, n = rem & (N_ - 1);
    const float* nw = which ? knw : qnw;
    __bf16* row = qkv + (((size_t)which * B_ + b) * H_ + h) * (size_t)N_ * DP_
                + (size_t)n * DP_;
    float y1[2], y2[2];
    float ss = 0.f;
    int cnt = 0;
    for (int i = lane; i < 36; i += 32) {
      float x1 = (float)row[i], x2 = (float)row[i + 36];
      float ang = (float)n * fr[cnt];
      float cs = __cosf(ang), sn = __sinf(ang);
      float a = x1 * cs - x2 * sn;
      float c = x2 * cs + x1 * sn;
      y1[cnt] = a; y2[cnt] = c; ++cnt;
      ss += a * a + c * c;
    }
#pragma unroll
    for (int off = 1; off < 32; off <<= 1) ss += __shfl_xor(ss, off, 32);
    float rn = rsqrtf(ss * (1.0f / 72.0f) + 1e-6f);
    cnt = 0;
    for (int i = lane; i < 36; i += 32) {
      row[i]      = (__bf16)(nw[i]      * (y1[cnt] * rn));
      row[i + 36] = (__bf16)(nw[i + 36] * (y2[cnt] * rn));
      ++cnt;
    }
  }
}

// ---------------------------------------------------------------------------
// Kernel 3: flash attention, 128 queries/block, 32-key chunks
// ---------------------------------------------------------------------------
__global__ __launch_bounds__(256)
void k_attn(const __bf16* __restrict__ qkv, __bf16* __restrict__ ctx) {
  __shared__ __bf16 sK[32][104];      // [key][dh]
  __shared__ __bf16 sVt[96][40];      // [dh][key] (transposed)
  __shared__ __bf16 sP[8][16][40];    // per-wave P staging (C/D -> A relayout)

  const int t = threadIdx.x, w = t >> 5, lane = t & 31;
  const int half = lane >> 4, l16 = lane & 15;
  const int bh = blockIdx.y, b = bh >> 4, h = bh & 15;
  const int qw = blockIdx.x * 128 + w * 16;

  const size_t plane = (size_t)N_ * DP_;
  const __bf16* Q = qkv + ((size_t)(0 * B_ + b) * H_ + h) * plane;
  const __bf16* K = qkv + ((size_t)(1 * B_ + b) * H_ + h) * plane;
  const __bf16* V = qkv + ((size_t)(2 * B_ + b) * H_ + h) * plane;

  v16bf aQ[3];
  {
    const __bf16* qb = Q + (size_t)(qw + l16) * DP_;
#pragma unroll
    for (int s = 0; s < 3; ++s) aQ[s] = frag_a(qb + s * 32, half);
  }

  v8f o[5];
#pragma unroll
  for (int i = 0; i < 5; ++i) o[i] = vzero8();
  float m8[8], l8[8];
#pragma unroll
  for (int r = 0; r < 8; ++r) { m8[r] = -1e30f; l8[r] = 0.f; }
  const float scale = 0.1178511301977579f;   // 72^-0.5

  for (int k0 = 0; k0 < N_; k0 += 32) {
#pragma unroll
    for (int it = 0; it < 2; ++it) {          // K chunk: 32x96 = 384 x b128
      int idx = t + it * 256;
      if (idx < 384) {
        int row = idx / 12, c8 = (idx % 12) * 8;
#if ASYNC_COPY
        async_b128(K + (size_t)(k0 + row) * DP_ + c8, &sK[row][c8]);
#else
        *(v8bf*)&sK[row][c8] = *(const v8bf*)(K + (size_t)(k0 + row) * DP_ + c8);
#endif
      }
    }
#pragma unroll
    for (int it = 0; it < 3; ++it) {          // V chunk transposed: 768 x b64
      int idx = t + it * 256;
      int row = idx / 24, c4 = (idx % 24) * 4;
      v4bf v = *(const v4bf*)(V + (size_t)(k0 + row) * DP_ + c4);
      sVt[c4 + 0][row] = v[0];
      sVt[c4 + 1][row] = v[1];
      sVt[c4 + 2][row] = v[2];
      sVt[c4 + 3][row] = v[3];
    }
#if ASYNC_COPY
    async_wait0();
#endif
    __syncthreads();

    v8f s0 = vzero8(), s1 = vzero8();
#pragma unroll
    for (int s = 0; s < 3; ++s) {
      v16bf bK0 = frag_b(&sK[l16][s * 32], half);
      v16bf bK1 = frag_b(&sK[16 + l16][s * 32], half);
      s0 = wmma_bf16(aQ[s], bK0, s0);
      s1 = wmma_bf16(aQ[s], bK1, s1);
    }

#pragma unroll
    for (int r = 0; r < 8; ++r) {
      float a = s0[r] * scale, c = s1[r] * scale;
      float mx = fmaxf(a, c);
#pragma unroll
      for (int off = 1; off < 16; off <<= 1)
        mx = fmaxf(mx, __shfl_xor(mx, off, 32));
      float mnew = fmaxf(m8[r], mx);
      float corr = __expf(m8[r] - mnew);
      float p0 = __expf(a - mnew);
      float p1 = __expf(c - mnew);
      float rs = p0 + p1;
#pragma unroll
      for (int off = 1; off < 16; off <<= 1) rs += __shfl_xor(rs, off, 32);
      l8[r] = l8[r] * corr + rs;
      m8[r] = mnew;
#pragma unroll
      for (int tt = 0; tt < 5; ++tt) o[tt][r] *= corr;
      sP[w][r + 8 * half][l16]      = (__bf16)p0;
      sP[w][r + 8 * half][16 + l16] = (__bf16)p1;
    }

    v16bf aP = frag_a(&sP[w][l16][0], half);
#pragma unroll
    for (int tt = 0; tt < 5; ++tt) {          // dh 80..95 is zero pad: skip
      v16bf bV = frag_b(&sVt[tt * 16 + l16][0], half);
      o[tt] = wmma_bf16(aP, bV, o[tt]);
    }
    __syncthreads();
  }

  float inv[8];
#pragma unroll
  for (int r = 0; r < 8; ++r) inv[r] = 1.0f / l8[r];
#pragma unroll
  for (int tt = 0; tt < 5; ++tt) {
    int dh = tt * 16 + l16;
    if (dh < DH_) {
#pragma unroll
      for (int r = 0; r < 8; ++r) {
        int row = qw + r + 8 * half;
        ctx[((size_t)b * N_ + row) * C_ + h * DH_ + dh] =
            (__bf16)(o[tt][r] * inv[r]);
      }
    }
  }
}

// ---------------------------------------------------------------------------
// Kernel 4: out = ctx @ w_proj^T + b_proj   (f32 output)
// ---------------------------------------------------------------------------
__global__ __launch_bounds__(256)
void k_proj(const __bf16* __restrict__ ctx, const float* __restrict__ wproj,
            const float* __restrict__ bproj, float* __restrict__ out) {
  __shared__ __bf16 sA[64][40];
  __shared__ __bf16 sB[128][40];
  const int t = threadIdx.x;
  const int m0 = blockIdx.x * 64;
  const int n0 = blockIdx.y * 128;
  const int w = t >> 5, lane = t & 31;
  const int half = lane >> 4, l16 = lane & 15;
  const int mw = (w & 1) * 32, nw = (w >> 1) * 32;

  v8f acc[2][2];
  acc[0][0] = vzero8(); acc[0][1] = vzero8();
  acc[1][0] = vzero8(); acc[1][1] = vzero8();

  for (int kt = 0; kt < C_; kt += 32) {
    {                                          // A already bf16: 256 x b128
      int row = t >> 2, c8 = (t & 3) * 8;
#if ASYNC_COPY
      async_b128(ctx + (size_t)(m0 + row) * C_ + kt + c8, &sA[row][c8]);
#else
      *(v8bf*)&sA[row][c8] = *(const v8bf*)(ctx + (size_t)(m0 + row) * C_ + kt + c8);
#endif
    }
#pragma unroll
    for (int it = 0; it < 4; ++it) {           // B: 128x32 f32 -> bf16
      int idx = t + it * 256;
      int row = idx >> 3, c4 = (idx & 7) * 4;
      float4 f = *(const float4*)(wproj + (size_t)(n0 + row) * C_ + kt + c4);
      v4bf v; v[0] = (__bf16)f.x; v[1] = (__bf16)f.y;
      v[2] = (__bf16)f.z; v[3] = (__bf16)f.w;
      *(v4bf*)&sB[row][c4] = v;
    }
    if (kt + 32 < C_) {                        // prefetch next k-tile
      __builtin_prefetch(wproj + (size_t)(n0 + (t >> 3)) * C_ + kt + 32, 0, 0);
    }
#if ASYNC_COPY
    async_wait0();
#endif
    __syncthreads();
    v16bf a0 = frag_a(&sA[mw + l16][0], half);
    v16bf a1 = frag_a(&sA[mw + 16 + l16][0], half);
    v16bf b0 = frag_b(&sB[nw + l16][0], half);
    v16bf b1 = frag_b(&sB[nw + 16 + l16][0], half);
    acc[0][0] = wmma_bf16(a0, b0, acc[0][0]);
    acc[0][1] = wmma_bf16(a0, b1, acc[0][1]);
    acc[1][0] = wmma_bf16(a1, b0, acc[1][0]);
    acc[1][1] = wmma_bf16(a1, b1, acc[1][1]);
    __syncthreads();
  }

#pragma unroll
  for (int i = 0; i < 2; ++i)
#pragma unroll
    for (int j = 0; j < 2; ++j)
#pragma unroll
      for (int r = 0; r < 8; ++r) {
        int gm = m0 + mw + i * 16 + r + 8 * half;
        int d  = n0 + nw + j * 16 + l16;
        out[(size_t)gm * C_ + d] = acc[i][j][r] + bproj[d];
      }
}

// ---------------------------------------------------------------------------
extern "C" void kernel_launch(void* const* d_in, const int* in_sizes, int n_in,
                              void* d_out, int out_size, void* d_ws, size_t ws_size,
                              hipStream_t stream) {
  (void)in_sizes; (void)n_in; (void)out_size; (void)ws_size;
  const float* x     = (const float*)d_in[0];
  const float* wqkv  = (const float*)d_in[1];
  const float* wproj = (const float*)d_in[2];
  const float* bproj = (const float*)d_in[3];
  const float* qnw   = (const float*)d_in[4];
  const float* knw   = (const float*)d_in[5];
  float* out = (float*)d_out;

  const size_t qkvBytes = (size_t)3 * B_ * H_ * N_ * DP_ * sizeof(__bf16);
  __bf16* qkv = (__bf16*)d_ws;
  __bf16* ctx = (__bf16*)((char*)d_ws + qkvBytes);

  (void)hipMemsetAsync(qkv, 0, qkvBytes, stream); // zero the DH->DP pads
  k_qkv_gemm<<<dim3(M_ / 64, CO_ / 128), 256, 0, stream>>>(x, wqkv, qkv);
  k_rope_rms<<<2048, 256, 0, stream>>>(qkv, qnw, knw);
  k_attn<<<dim3(N_ / 128, B_ * H_), 256, 0, stream>>>(qkv, ctx);
  k_proj<<<dim3(M_ / 64, C_ / 128), 256, 0, stream>>>(ctx, wproj, bproj, out);
}
